// GraphAttentionLayer_36532991820135
// MI455X (gfx1250) — matrix operand order repaired and checked by reading, compiled, not run on hardware
//
#include <hip/hip_runtime.h>
#include <hip/hip_bf16.h>
#include <hip/hip_fp16.h>

// GAT layer for MI455X (gfx1250, wave32, WMMA).
// Flash-attention style streaming over the 8192x8192 attention matrix.
// HBM floor = adj read (256 MB) ~= 11us @ 23.3 TB/s; h_t (1 MB) is L2-resident.
// Matmuls use v_wmma_f32_16x16x32_f16; h^T tile staging uses the gfx1250
// async global->LDS copy path (ASYNCcnt) when the toolchain exposes it.

typedef __attribute__((ext_vector_type(16))) _Float16 v16h;
typedef __attribute__((ext_vector_type(8)))  float    v8f;
typedef __attribute__((ext_vector_type(4)))  int      v4i_;

#define N_NODES 8192
#define FIN     512
#define FOUT    64
#define NEGV    (-9e15f)
#define SLOPE   0.2f

#if __has_builtin(__builtin_amdgcn_global_load_async_to_lds_b128) && \
    __has_builtin(__builtin_amdgcn_s_wait_asynccnt)
#define HAVE_ASYNC_LDS 1
typedef __attribute__((address_space(1))) v4i_* async_gptr;
typedef __attribute__((address_space(3))) v4i_* async_lptr;
#else
#define HAVE_ASYNC_LDS 0
#endif

union FragU  { v16h h; unsigned int u[8]; };
union Pack8U { uint4 v; _Float16 h[8]; };

// ---------------------------------------------------------------------------
// Kernel 1: h = x @ W + b. Stores h_f32 (for score dots) and a TRANSPOSED
// f16 copy h_t[64][8192] so the attention kernel's B-tile staging is pure
// vector copies (no per-iteration transpose).
// ---------------------------------------------------------------------------
__global__ void __launch_bounds__(128) gat_h_kernel(
    const float* __restrict__ x, const float* __restrict__ W,
    const float* __restrict__ b, float* __restrict__ h_f32,
    _Float16* __restrict__ h_t) {
  __shared__ __align__(16) _Float16 lds_x[16 * 32];   // A tile, row-major [m][k]
  __shared__ __align__(16) _Float16 lds_wt[64 * 32];  // B tile transposed [n][k]

  const int tid  = threadIdx.x;
  const int lane = tid & 31;
  const int wave = tid >> 5;
  const int mb   = blockIdx.x * 16;

  v8f c = {};

  for (int k0 = 0; k0 < FIN; k0 += 32) {
    for (int q = tid; q < 16 * 32; q += 128) {
      int r = q >> 5, cc = q & 31;
      lds_x[q] = (_Float16)x[(mb + r) * FIN + k0 + cc];
    }
    for (int q = tid; q < 32 * 64; q += 128) {
      int k = q >> 6, n = q & 63;
      lds_wt[n * 32 + k] = (_Float16)W[(k0 + k) * FOUT + n];
    }
    __syncthreads();

    // WMMA fragments per CDNA5 ISA 7.12.2 wave32 layouts.
    FragU fa, fb;
    const unsigned int* xu = (const unsigned int*)lds_x;   // [16][16] u32 pairs
    const unsigned int* wu = (const unsigned int*)lds_wt;  // [64][16] u32 pairs
    const int m   = lane & 15;
    const int hi4 = (lane >> 4) << 2;
    const int hi8 = (lane >> 4) << 3;
#pragma unroll
    for (int v = 0; v < 8; ++v)
      fa.u[v] = xu[m * 16 + v + ((v >> 2) << 2) + hi4];
    const int n = (wave << 4) + (lane & 15);
#pragma unroll
    for (int v = 0; v < 8; ++v)
      fb.u[v] = wu[n * 16 + v + ((v >> 2) << 2) + hi4];

    c = __builtin_amdgcn_wmma_f32_16x16x32_f16(
        false, fa.h, false, fb.h, (short)0, c, false, false);
    __syncthreads();
  }

  // C/D layout: lane, VGPR r -> M = r + (lane<16?0:8), N = lane%16.
  const int n    = (wave << 4) + (lane & 15);
  const int moff = (lane >> 4) << 3;
  const float bias = b[n];
#pragma unroll
  for (int r = 0; r < 8; ++r) {
    int row = mb + r + moff;
    float hv = c[r] + bias;
    h_f32[row * FOUT + n] = hv;
    h_t[(size_t)n * N_NODES + row] = (_Float16)hv;  // transposed copy
  }
}

// ---------------------------------------------------------------------------
// Kernel 2: s_src[i] = h[i].a[:64], s_dst[i] = h[i].a[64:]
// ---------------------------------------------------------------------------
__global__ void gat_scores_kernel(const float* __restrict__ h,
                                  const float* __restrict__ a,
                                  float* __restrict__ s_src,
                                  float* __restrict__ s_dst) {
  int i = blockIdx.x * blockDim.x + threadIdx.x;
  if (i >= N_NODES) return;
  float ss = 0.f, sd = 0.f;
#pragma unroll 8
  for (int f = 0; f < FOUT; ++f) {
    float hv = h[i * FOUT + f];
    ss += hv * a[f];
    sd += hv * a[FOUT + f];
  }
  s_src[i] = ss;
  s_dst[i] = sd;
}

// ---------------------------------------------------------------------------
// Kernel 3: streaming masked softmax + (attn @ h) + ELU.
// Block = 256 threads (8 waves) owns 32 rows; stream 64 columns per step.
// Wave w: m-subtile = (w>>2)*16, n-tile = (w&3)*16.
// ---------------------------------------------------------------------------
__global__ void __launch_bounds__(256) gat_attn_kernel(
    const int* __restrict__ adj, const float* __restrict__ s_src,
    const float* __restrict__ s_dst, const float* __restrict__ a_b_p,
    const _Float16* __restrict__ h_t, float* __restrict__ out) {
  __shared__ __align__(16) float    e_lds[32 * 64];    // energies (8 KB)
  __shared__ __align__(16) _Float16 p_lds[32 * 64];    // P tile   (4 KB)
  __shared__ __align__(16) _Float16 ht_lds[64 * 64];   // h^T tile (8 KB)
  __shared__ float m_lds[32], l_lds[32], alpha_lds[32], srow[32], tsum[32];

  const int tid  = threadIdx.x;
  const int lane = tid & 31;
  const int wave = tid >> 5;
  const int mb   = blockIdx.x * 32;
  const float ab = a_b_p[0];

  if (tid < 32) {
    m_lds[tid] = -INFINITY;
    l_lds[tid] = 0.f;
    srow[tid]  = s_src[mb + tid];
  }
  v8f c = {};
  __syncthreads();

  // Energy ownership: 8 contiguous columns of one row of the 32x64 tile.
  const int erow = tid >> 3;         // 0..31
  const int ejj  = (tid & 7) << 3;   // 0,8,...,56
  // h^T staging ownership: 16 contiguous k of one n-row of the 64x64 tile.
  const int sn = tid >> 2;           // 0..63
  const int sk = (tid & 3) << 4;     // 0,16,32,48

  const int msub = (wave >> 2) << 4; // 0 or 16
  const int ntl  = (wave & 3) << 4;  // 0,16,32,48

  float ev[8];  // energies carried in registers from phase A to phase C

  for (int j0 = 0; j0 < N_NODES; j0 += 64) {
    // --- Phase A: stage h^T tile; masked energies --------------------------
    if (tid < 32) tsum[tid] = 0.f;
    {
      const _Float16* gsrc = h_t + (size_t)sn * N_NODES + j0 + sk;
      _Float16*       ldst = &ht_lds[sn * 64 + sk];
#if HAVE_ASYNC_LDS
      // gfx1250 async copy: 16B per lane per op; imm offset applies to both
      // global and LDS addresses (ISA async addressing). ASYNCcnt-tracked.
      async_gptr gp = (async_gptr)gsrc;  // C-style cast: drops const + addrspace
      async_lptr lp = (async_lptr)ldst;
      __builtin_amdgcn_global_load_async_to_lds_b128(gp, lp, 0, 0);
      __builtin_amdgcn_global_load_async_to_lds_b128(gp, lp, 16, 0);
#else
      const uint4* hp = (const uint4*)gsrc;
      uint4 q0 = hp[0], q1 = hp[1];
      ((uint4*)ldst)[0] = q0;
      ((uint4*)ldst)[1] = q1;
#endif
    }
    {
      const int* arow = adj + (mb + erow) * N_NODES + j0 + ejj;
      int4   a0 = ((const int4*)arow)[0];
      int4   a1 = ((const int4*)arow)[1];
      float4 d0 = ((const float4*)(s_dst + j0 + ejj))[0];
      float4 d1 = ((const float4*)(s_dst + j0 + ejj))[1];
      const float si = srow[erow];
      float dv[8] = {d0.x, d0.y, d0.z, d0.w, d1.x, d1.y, d1.z, d1.w};
      int   av[8] = {a0.x, a0.y, a0.z, a0.w, a1.x, a1.y, a1.z, a1.w};
#pragma unroll
      for (int v = 0; v < 8; ++v) {
        float e = si + dv[v] + ab;
        e = (e > 0.f) ? e : SLOPE * e;
        ev[v] = (av[v] != 0) ? e : NEGV;
      }
      float4* ep = (float4*)&e_lds[erow * 64 + ejj];
      ep[0] = make_float4(ev[0], ev[1], ev[2], ev[3]);
      ep[1] = make_float4(ev[4], ev[5], ev[6], ev[7]);
      // Prefetch next adj tile (gfx1250 global_prefetch); uniform guard.
      if (j0 + 64 < N_NODES) __builtin_prefetch(arow + 64, 0, 1);
    }
#if HAVE_ASYNC_LDS
    __builtin_amdgcn_s_wait_asynccnt(0);
#endif
    __syncthreads();

    // --- Phase B: per-row running max + rescale factor (wave 0, 32 lanes) --
    if (wave == 0) {
      float mx = m_lds[lane];
      float tm = -INFINITY;
      const float4* er = (const float4*)&e_lds[lane * 64];
#pragma unroll
      for (int v4 = 0; v4 < 16; ++v4) {
        float4 e4 = er[v4];
        tm = fmaxf(tm, fmaxf(fmaxf(e4.x, e4.y), fmaxf(e4.z, e4.w)));
      }
      float nm = fmaxf(mx, tm);
      alpha_lds[lane] = __expf(mx - nm);  // 0 on first tile (mx = -inf)
      m_lds[lane] = nm;
    }
    __syncthreads();

    // --- Phase C: P = exp(e - m) from registers; b128 store + one atomic ---
    {
      const float mi = m_lds[erow];
      Pack8U pk;
      float psum = 0.f;
#pragma unroll
      for (int v = 0; v < 8; ++v) {
        float p = __expf(ev[v] - mi);
        psum += p;
        pk.h[v] = (_Float16)p;
      }
      *(uint4*)&p_lds[erow * 64 + ejj] = pk.v;
      atomicAdd(&tsum[erow], psum);
    }
    __syncthreads();

    // --- Phase D: l update; rescale accumulators; 2x WMMA (K=64) -----------
    if (wave == 0)
      l_lds[lane] = l_lds[lane] * alpha_lds[lane] + tsum[lane];
    {
      const int moff = (lane >> 4) << 3;
#pragma unroll
      for (int r = 0; r < 8; ++r) c[r] *= alpha_lds[msub + r + moff];

      const unsigned int* pu = (const unsigned int*)p_lds;   // [32][32] pairs
      const unsigned int* hu = (const unsigned int*)ht_lds;  // [64][32] pairs
      const int m   = lane & 15;
      const int hi4 = (lane >> 4) << 2;
      const int hi8 = (lane >> 4) << 3;
      const int n   = ntl + (lane & 15);
#pragma unroll
      for (int kc = 0; kc < 2; ++kc) {
        FragU fp, fh;
#pragma unroll
        for (int v = 0; v < 8; ++v)
          fp.u[v] = pu[(msub + m) * 32 + kc * 16 + v + ((v >> 2) << 2) + hi4];
#pragma unroll
        for (int v = 0; v < 8; ++v)
          fh.u[v] = hu[n * 32 + kc * 16 + v + ((v >> 2) << 2) + hi4];
        c = __builtin_amdgcn_wmma_f32_16x16x32_f16(
            false, fp.h, false, fh.h, (short)0, c, false, false);
      }
    }
    __syncthreads();  // protect LDS tiles + alpha for next iteration
  }

  // Epilogue: h_prime = acc / l, then ELU.
  const int n    = ntl + (lane & 15);
  const int moff = (lane >> 4) << 3;
#pragma unroll
  for (int r = 0; r < 8; ++r) {
    int row = mb + msub + r + moff;
    float v = c[r] / l_lds[msub + r + moff];
    v = (v > 0.f) ? v : (__expf(v) - 1.f);
    out[row * FOUT + n] = v;
  }
}

// ---------------------------------------------------------------------------
// Host launch
// ---------------------------------------------------------------------------
extern "C" void kernel_launch(void* const* d_in, const int* in_sizes, int n_in,
                              void* d_out, int out_size, void* d_ws, size_t ws_size,
                              hipStream_t stream) {
  (void)in_sizes; (void)n_in; (void)out_size; (void)ws_size;
  const float* x   = (const float*)d_in[0];
  const int*   adj = (const int*)d_in[1];
  const float* W   = (const float*)d_in[2];
  const float* b   = (const float*)d_in[3];
  const float* a   = (const float*)d_in[4];
  const float* a_b = (const float*)d_in[5];
  float* out = (float*)d_out;

  // Workspace layout
  char* ws = (char*)d_ws;
  float*    h_f32 = (float*)ws;                                    // 2 MB
  _Float16* h_t   = (_Float16*)(ws + (size_t)N_NODES * FOUT * 4);  // 1 MB (transposed)
  float*    s_src = (float*)(ws + (size_t)N_NODES * FOUT * 6);
  float*    s_dst = s_src + N_NODES;

  gat_h_kernel<<<N_NODES / 16, 128, 0, stream>>>(x, W, b, h_f32, h_t);
  gat_scores_kernel<<<N_NODES / 256, 256, 0, stream>>>(h_f32, a, s_src, s_dst);
  gat_attn_kernel<<<N_NODES / 32, 256, 0, stream>>>(adj, s_src, s_dst, a_b,
                                                    h_t, out);
}
